// MSDeformAttn_3100966388123
// MI455X (gfx1250) — matrix operand order repaired and checked by reading, not compiled, not run on hardware
//
#include <hip/hip_runtime.h>

// MS-Deformable-Attention for MI455X (gfx1250, wave32, WMMA).
// f32->f16 once | software-pipelined WMMA GEMMs (16x128/wave, double-buffered
// fragments) | fused softmax+bilinear gather (f16 value cache ~89MB ->
// L2-resident on 192MB L2) | WMMA out-projection.

typedef __attribute__((ext_vector_type(16))) _Float16 v16h;
typedef __attribute__((ext_vector_type(8)))  _Float16 v8h;
typedef __attribute__((ext_vector_type(8)))  float    v8f;

#define DMODEL 256
#define NHEAD  8
#define CDIM   32

// ---------- fragment loaders -------------------------------------------------
// A (16x32 f16): lane half g=lane>>4; elems 0..15 <- K = k+g*16 .. +15
//                (contiguous, matching B)
__device__ __forceinline__ v16h load_a_frag(const _Float16* p) {
  const v8h* q = (const v8h*)p;        // 16B-aligned: k%32==0, g*16 halfs = 32B
  v8h a0 = q[0], a1 = q[1];            // +8 halfs
  v16h f;
#pragma unroll
  for (int i = 0; i < 8; ++i) { f[i] = a0[i]; f[8 + i] = a1[i]; }
  return f;
}

// B (32x16 f16): lane half holds 16 contiguous K (this code's order, same as A):
// p = WT[n]*K + k + g*16, read 16 contiguous halfs.
__device__ __forceinline__ v16h load_b_frag(const _Float16* p) {
  const v8h* q = (const v8h*)p;        // 16B-aligned: k%32==0, g*16 halfs = 32B
  v8h b0 = q[0], b1 = q[1];
  v16h f;
#pragma unroll
  for (int i = 0; i < 8; ++i) { f[i] = b0[i]; f[8 + i] = b1[i]; }
  return f;
}

// ---------- elementwise f32 -> f16 (8 elems/thread) --------------------------
__global__ void __launch_bounds__(256) cvt_f16x8(const float* __restrict__ in,
                                                 _Float16* __restrict__ out) {
  const size_t i = ((size_t)blockIdx.x * 256 + threadIdx.x) * 8;
  const float4* p = (const float4*)(in + i);
  float4 a = p[0], b = p[1];
  v8h o;
  o[0]=(_Float16)a.x; o[1]=(_Float16)a.y; o[2]=(_Float16)a.z; o[3]=(_Float16)a.w;
  o[4]=(_Float16)b.x; o[5]=(_Float16)b.y; o[6]=(_Float16)b.z; o[7]=(_Float16)b.w;
  *(v8h*)(out + i) = o;
}

// ---------- weight transpose + f32->f16 -------------------------------------
__global__ void __launch_bounds__(256) transpose_cvt(const float* __restrict__ W,
                                                     _Float16* __restrict__ WT,
                                                     int K, int N) {
  int idx = blockIdx.x * 256 + threadIdx.x;  // grid sized exactly to K*N
  int k = idx / N, n = idx - k * N;
  WT[(size_t)n * K + k] = (_Float16)W[idx];
}

// ---------- WMMA GEMM: C[M,N] = A[M,K] * WT[N,K]^T + bias --------------------
// one wave -> 16x128 tile (8 accumulators). Software pipelined: next K-step's
// A/B fragments are loaded before the current step's 8 WMMAs, so vmem latency
// overlaps the tensor pipe instead of s_wait_loadcnt 0x0 before every WMMA.
template <typename OutT>
__global__ void __launch_bounds__(256) gemm_wmma(const _Float16* __restrict__ A,
                                                 const _Float16* __restrict__ WT,
                                                 const float* __restrict__ bias,
                                                 OutT* __restrict__ C,
                                                 int M, int N, int K) {
  const int lane  = threadIdx.x & 31;
  const int wave  = blockIdx.x * 8 + (threadIdx.x >> 5);
  const int ntile = N >> 7;                  // 128-wide per wave
  const int mt = wave / ntile;
  const int nt = wave - mt * ntile;
  const int m0 = mt << 4;
  const int n0 = nt << 7;
  const int grp = lane >> 4;
  const int l16 = lane & 15;

  const _Float16* arow = A  + (size_t)(m0 + l16) * K + grp * 16;
  const _Float16* brow = WT + (size_t)(n0 + l16) * K + grp * 16;

  v8f acc[8] = {};

  // prologue: load K-step 0 fragments
  v16h a_cur = load_a_frag(arow);
  v16h b_cur[8];
#pragma unroll
  for (int j = 0; j < 8; ++j) b_cur[j] = load_b_frag(brow + (size_t)j * 16 * K);

  for (int k = 0; k < K - 32; k += 32) {
    // issue next K-step loads first (overlap with WMMAs below)
    v16h a_nxt = load_a_frag(arow + k + 32);
    v16h b_nxt[8];
#pragma unroll
    for (int j = 0; j < 8; ++j)
      b_nxt[j] = load_b_frag(brow + k + 32 + (size_t)j * 16 * K);

#pragma unroll
    for (int j = 0; j < 8; ++j)
      acc[j] = __builtin_amdgcn_wmma_f32_16x16x32_f16(
          false, a_cur, false, b_cur[j], (short)0, acc[j], false, false);

    a_cur = a_nxt;
#pragma unroll
    for (int j = 0; j < 8; ++j) b_cur[j] = b_nxt[j];
  }
  // epilogue K-step
#pragma unroll
  for (int j = 0; j < 8; ++j)
    acc[j] = __builtin_amdgcn_wmma_f32_16x16x32_f16(
        false, a_cur, false, b_cur[j], (short)0, acc[j], false, false);

  // C/D layout: lane half g, VGPR r -> M = m0 + g*8 + r, N = n0 + j*16 + l16
#pragma unroll
  for (int j = 0; j < 8; ++j) {
    const int n = n0 + j * 16 + l16;
    const float bv = bias[n];
#pragma unroll
    for (int r = 0; r < 8; ++r) {
      const int m = m0 + (grp << 3) + r;
      C[(size_t)m * N + n] = (OutT)(acc[j][r] + bv);
    }
  }
}

// ---------- fused softmax + deformable bilinear gather -----------------------
// one wave per (b,q,head); lane = channel. v is f16 (L2-resident ~89MB).
__global__ void __launch_bounds__(256) deform_gather(const _Float16* __restrict__ v,
                                                     const float* __restrict__ offb,
                                                     const float* __restrict__ refer,
                                                     const float* __restrict__ logits,
                                                     const int* __restrict__ shapes,
                                                     _Float16* __restrict__ accout,
                                                     int Q, int LV) {
  const int lane = threadIdx.x & 31;
  const int gw   = blockIdx.x * 8 + (threadIdx.x >> 5);
  const int h    = gw & 7;
  const int bq   = gw >> 3;
  const int b    = bq / Q;

  const float* offrow = offb   + (size_t)bq * 256 + h * 32;  // [l][p][xy]
  const float* lrow   = logits + (size_t)bq * 128 + h * 16;  // [l][p]
  const float* refrow = refer  + (size_t)bq * 8;             // [l][xy]
  const _Float16* vbase = v + (size_t)b * LV * DMODEL + h * CDIM + lane;

  // wave-uniform softmax over the 16 (level,point) logits (redundant per lane)
  float w16[16];
  float mx = lrow[0];
#pragma unroll
  for (int i = 1; i < 16; ++i) mx = fmaxf(mx, lrow[i]);
  float s = 0.f;
#pragma unroll
  for (int i = 0; i < 16; ++i) { w16[i] = __expf(lrow[i] - mx); s += w16[i]; }
  const float inv = 1.f / s;
#pragma unroll
  for (int i = 0; i < 16; ++i) w16[i] *= inv;

  float accv = 0.f;
  int start = 0;
#pragma unroll
  for (int l = 0; l < 4; ++l) {
    const int H = shapes[l * 2 + 0];
    const int W = shapes[l * 2 + 1];
    const float rx = refrow[l * 2 + 0] * (float)W - 0.5f;
    const float ry = refrow[l * 2 + 1] * (float)H - 0.5f;
    const _Float16* img = vbase + (size_t)start * DMODEL;
#pragma unroll
    for (int p = 0; p < 4; ++p) {
      // loc*W - 0.5 == r*W + off - 0.5  (norm cancels)
      const float x  = rx + offrow[l * 8 + p * 2 + 0];
      const float y  = ry + offrow[l * 8 + p * 2 + 1];
      const float xf = floorf(x), yf = floorf(y);
      const float fx = x - xf,    fy = y - yf;
      const int   x0 = (int)xf,   y0 = (int)yf;
      const float aw = w16[l * 4 + p];
      float smp = 0.f;
#pragma unroll
      for (int dy = 0; dy < 2; ++dy) {
        const int yi = y0 + dy;
        if (yi < 0 || yi >= H) continue;          // wave-uniform condition
        const float wy = dy ? fy : 1.f - fy;
#pragma unroll
        for (int dx = 0; dx < 2; ++dx) {
          const int xi = x0 + dx;
          if (xi < 0 || xi >= W) continue;
          const float wx = dx ? fx : 1.f - fx;
          smp += wy * wx * (float)img[(size_t)(yi * W + xi) * DMODEL]; // 64B coalesced
        }
      }
      accv += aw * smp;
    }
    start += H * W;
  }
  accout[(size_t)bq * 256 + h * CDIM + lane] = (_Float16)accv;
}

// ---------- host orchestration ----------------------------------------------
extern "C" void kernel_launch(void* const* d_in, const int* in_sizes, int n_in,
                              void* d_out, int out_size, void* d_ws, size_t ws_size,
                              hipStream_t stream) {
  (void)n_in; (void)out_size; (void)ws_size;
  const float* query = (const float*)d_in[0];
  const float* refer = (const float*)d_in[1];
  const float* value = (const float*)d_in[2];
  const int*   shapes= (const int*)  d_in[3];
  const float* Wv = (const float*)d_in[4];
  const float* bv = (const float*)d_in[5];
  const float* Wo = (const float*)d_in[6];
  const float* bo = (const float*)d_in[7];
  const float* Wa = (const float*)d_in[8];
  const float* ba = (const float*)d_in[9];
  const float* Wu = (const float*)d_in[10];
  const float* bu = (const float*)d_in[11];
  float* out = (float*)d_out;

  const int MV = in_sizes[2] / DMODEL;   // bs*LV = 174080
  const int MQ = in_sizes[0] / DMODEL;   // bs*Q  = 16384
  const int LV = MV / 8;                 // bs = 8
  const int Q  = MQ / 8;

  // workspace carve-up (ws from hipMalloc -> 256B aligned; all offsets 16B)
  _Float16* wtv = (_Float16*)d_ws;                  // 256x256
  _Float16* wto = wtv + 256 * 256;                  // 256x256
  _Float16* wta = wto + 256 * 256;                  // 128x256
  _Float16* wtu = wta + 128 * 256;                  // 256x256
  _Float16* valh   = wtu + 256 * 256;               // MV x 256 (f16 value)
  _Float16* qh     = valh + (size_t)MV * 256;       // MQ x 256 (f16 query)
  _Float16* vprojh = qh   + (size_t)MQ * 256;       // MV x 256 (f16 v-proj)
  _Float16* acch   = vprojh + (size_t)MV * 256;     // MQ x 256 (f16 gather out)
  float* offbuf  = (float*)(acch + (size_t)MQ * 256);   // MQ x 256
  float* logitb  = offbuf + (size_t)MQ * 256;           // MQ x 128 (raw logits)

  // 1) inputs -> f16 once (halves GEMM A traffic, removes inner-loop cvt)
  cvt_f16x8<<<MV / 8, 256, 0, stream>>>(value, valh);
  cvt_f16x8<<<MQ / 8, 256, 0, stream>>>(query, qh);

  // 2) weights -> f16, transposed (N-major, contiguous K)
  transpose_cvt<<<256, 256, 0, stream>>>(Wv, wtv, 256, 256);
  transpose_cvt<<<256, 256, 0, stream>>>(Wo, wto, 256, 256);
  transpose_cvt<<<128, 256, 0, stream>>>(Wa, wta, 256, 128);
  transpose_cvt<<<256, 256, 0, stream>>>(Wu, wtu, 256, 256);

  // 3) projections (grids exact: waves = (M/16)*(N/128), 8 waves/block)
  gemm_wmma<_Float16><<<(MV / 16) * 2 / 8, 256, 0, stream>>>(valh, wtv, bv, vprojh, MV, 256, 256);
  gemm_wmma<float>   <<<(MQ / 16) * 2 / 8, 256, 0, stream>>>(qh,   wto, bo, offbuf, MQ, 256, 256);
  gemm_wmma<float>   <<<(MQ / 16) * 1 / 8, 256, 0, stream>>>(qh,   wta, ba, logitb, MQ, 128, 256);

  // 4) fused softmax + bilinear gather over f16 value cache
  deform_gather<<<MQ, 256, 0, stream>>>(vprojh, offbuf, refer, logitb, shapes,
                                        acch, Q, LV);

  // 5) output projection -> d_out (f32)
  gemm_wmma<float><<<(MQ / 16) * 2 / 8, 256, 0, stream>>>(acch, wtu, bu, out, MQ, 256, 256);
}
